// SlotAttention_5712306503852
// MI455X (gfx1250) — compile-verified, matrix operand order checked by reading, not run on hardware
//
#include <hip/hip_runtime.h>
#include <stdint.h>

// ---------------------------------------------------------------------------
// CDNA5 (gfx1250) slot-attention. wave32, WMMA bf16 16x16x32, async-to-LDS.
// All GEMMs are A[MxK] @ B[NxK]^T with both operands contiguous in K.
// ---------------------------------------------------------------------------

typedef __bf16 bf16_t;
typedef __attribute__((ext_vector_type(16))) __bf16 v16bf;
typedef __attribute__((ext_vector_type(8)))  __bf16 v8bf;
typedef __attribute__((ext_vector_type(8)))  float  v8f;

union Frag16 { v16bf v; struct { v8bf lo, hi; } h; };

__device__ __forceinline__ bf16_t f2bf(float f) {
  union { float f; uint32_t u; } c; c.f = f;
  uint32_t r = (c.u + 0x7FFFu + ((c.u >> 16) & 1u)) >> 16;   // round-nearest-even
  union { uint16_t u; bf16_t b; } o; o.u = (uint16_t)r;
  return o.b;
}
__device__ __forceinline__ float sigmoidf_(float x) { return 1.f / (1.f + __expf(-x)); }

// ---------------------------------------------------------------------------
// f32 -> bf16 convert (grid-stride)
// ---------------------------------------------------------------------------
__global__ void f2bf_kernel(const float* __restrict__ src, bf16_t* __restrict__ dst, long n) {
  for (long i = (long)blockIdx.x * blockDim.x + threadIdx.x; i < n; i += (long)gridDim.x * blockDim.x)
    dst[i] = f2bf(src[i]);
}

// ---------------------------------------------------------------------------
// Tiled transpose + convert: src f32 (B, J=4096, D=256) -> dst bf16 (B, D, J)
// 32x32 tiles through LDS; grid = (J/32, D/32, B), block = 256.
// ---------------------------------------------------------------------------
__global__ void transpose_bf16_kernel(const float* __restrict__ src, bf16_t* __restrict__ dst) {
  __shared__ float tile[32][33];
  const int b = blockIdx.z;
  const int j0 = blockIdx.x * 32, d0 = blockIdx.y * 32;
  const int tx = threadIdx.x & 31, ty = threadIdx.x >> 5;   // 8 rows per pass
#pragma unroll
  for (int r = ty; r < 32; r += 8)
    tile[r][tx] = src[((long)b * 4096 + j0 + r) * 256 + d0 + tx];
  __syncthreads();
#pragma unroll
  for (int r = ty; r < 32; r += 8)
    dst[((long)b * 256 + d0 + r) * 4096 + j0 + tx] = f2bf(tile[tx][r]);
}

// ---------------------------------------------------------------------------
// slots = mu + sigma * noise   (B*NS*D elements, D=256)
// ---------------------------------------------------------------------------
__global__ void slots_init_kernel(const float* __restrict__ mu, const float* __restrict__ sigma,
                                  const float* __restrict__ noise, float* __restrict__ slots) {
  long id = (long)blockIdx.x * 256 + threadIdx.x;
  int d = id & 255;
  slots[id] = mu[d] + sigma[d] * noise[id];
}

// ---------------------------------------------------------------------------
// Row layernorm, 256 cols, one row per wave (8 rows / block of 256 threads).
// Row staged to LDS via CDNA5 async-to-LDS (ASYNCcnt), then wave-reduced.
// ---------------------------------------------------------------------------
__global__ void layernorm_rows_kernel(const float* __restrict__ x,
                                      const float* __restrict__ w,
                                      const float* __restrict__ b,
                                      bf16_t* __restrict__ out) {
  __shared__ float tile[8 * 256];
  const int wave = threadIdx.x >> 5;
  const int lane = threadIdx.x & 31;
  const long row = (long)blockIdx.x * 8 + wave;
  const float* src = x + row * 256;
  float* dst = &tile[wave * 256];

  {
    unsigned lds0 = (unsigned)(unsigned long long)(dst + lane * 4);
    unsigned long long g0 = (unsigned long long)(src + lane * 4);
    asm volatile("global_load_async_to_lds_b128 %0, %1, off" :: "v"(lds0), "v"(g0) : "memory");
    unsigned lds1 = (unsigned)(unsigned long long)(dst + 128 + lane * 4);
    unsigned long long g1 = (unsigned long long)(src + 128 + lane * 4);
    asm volatile("global_load_async_to_lds_b128 %0, %1, off" :: "v"(lds1), "v"(g1) : "memory");
    asm volatile("s_wait_asynccnt 0" ::: "memory");
  }

  float v[8], s = 0.f;
#pragma unroll
  for (int i = 0; i < 8; i++) { v[i] = dst[lane + 32 * i]; s += v[i]; }
#pragma unroll
  for (int off = 16; off > 0; off >>= 1) s += __shfl_xor(s, off, 32);
  const float mean = s * (1.f / 256.f);
  float q = 0.f;
#pragma unroll
  for (int i = 0; i < 8; i++) { float d = v[i] - mean; v[i] = d; q += d * d; }
#pragma unroll
  for (int off = 16; off > 0; off >>= 1) q += __shfl_xor(q, off, 32);
  const float rstd = rsqrtf(q * (1.f / 256.f) + 1e-5f);

  bf16_t* o = out + row * 256;
#pragma unroll
  for (int i = 0; i < 8; i++) {
    int c = lane + 32 * i;
    o[c] = f2bf(v[i] * rstd * w[c] + b[c]);
  }
}

// ---------------------------------------------------------------------------
// Batched WMMA GEMM:  C = alpha * A @ B^T + bias (+C if accum)
//   A : [M x K] bf16 row-major, batch stride sA
//   B : [N x K] bf16 row-major, batch stride sB
//   Block = 256 threads = 8 waves; block tile = 16 rows x 256 cols
//   (each wave: 2 adjacent 16x16 tiles sharing one A fragment).
//   grid = (M/16, N/256, batch). Stores guarded by Mvalid (M padding).
//   cbt=1: store bf16 transposed per 4096-row batch -> (m/4096, n, m%4096).
// ---------------------------------------------------------------------------
__global__ void gemm_wmma_kernel(const bf16_t* __restrict__ A, long sA,
                                 const bf16_t* __restrict__ B, long sB,
                                 const float* __restrict__ bias, float alpha,
                                 float* Cf, long sCf, int accum,
                                 bf16_t* Cb, long sCb, int cbt,
                                 int Mvalid, int N, int K, int relu) {
  const int bz = blockIdx.z;
  const bf16_t* Ab = A + (long)bz * sA;
  const bf16_t* Bb = B + (long)bz * sB;
  const int wave = threadIdx.x >> 5;
  const int lane = threadIdx.x & 31;
  const int mbase = blockIdx.x * 16;
  const int nbase = blockIdx.y * 256 + wave * 32;
  const int m  = lane & 15;            // A row / B col within tile
  const int kh = (lane >> 4) * 8;      // K-half select per documented layout

  v8f acc0 = {0.f, 0.f, 0.f, 0.f, 0.f, 0.f, 0.f, 0.f};
  v8f acc1 = {0.f, 0.f, 0.f, 0.f, 0.f, 0.f, 0.f, 0.f};

  const bf16_t* ap  = Ab + (long)(mbase + m) * K + kh;
  const bf16_t* bp0 = Bb + (long)(nbase + m) * K + kh;
  const bf16_t* bp1 = bp0 + (long)16 * K;

#pragma unroll 2
  for (int k0 = 0; k0 < K; k0 += 32) {
    Frag16 af, bf0, bf1;
    af.h.lo  = *(const v8bf*)(ap + k0);
    af.h.hi  = *(const v8bf*)(ap + k0 + 16);
    bf0.h.lo = *(const v8bf*)(bp0 + k0);
    bf0.h.hi = *(const v8bf*)(bp0 + k0 + 16);
    bf1.h.lo = *(const v8bf*)(bp1 + k0);
    bf1.h.hi = *(const v8bf*)(bp1 + k0 + 16);
    acc0 = __builtin_amdgcn_wmma_f32_16x16x32_bf16(false, af.v, false, bf0.v,
                                                   (short)0, acc0, false, false);
    acc1 = __builtin_amdgcn_wmma_f32_16x16x32_bf16(false, af.v, false, bf1.v,
                                                   (short)0, acc1, false, false);
  }

  // Epilogue (C/D layout: lane n = l&15; VGPR r -> M = 8*(l>>4)+r)
  const int rbase = (lane >> 4) * 8;
#pragma unroll
  for (int t = 0; t < 2; t++) {
    const v8f acc = t ? acc1 : acc0;
    const int n = nbase + t * 16 + (lane & 15);
    const float bvv = bias ? bias[n] : 0.f;
#pragma unroll
    for (int r = 0; r < 8; r++) {
      int mm = mbase + rbase + r;
      if (mm >= Mvalid) continue;
      float val = alpha * acc[r] + bvv;
      long ci = (long)mm * N + n;
      if (accum && Cf) val += Cf[(long)bz * sCf + ci];
      if (relu) val = fmaxf(val, 0.f);
      if (Cf) Cf[(long)bz * sCf + ci] = val;
      if (Cb) {
        long idx;
        if (cbt) { long bb = mm >> 12, jj = mm & 4095; idx = ((bb << 8) + n) * 4096 + jj; }
        else     { idx = (long)bz * sCb + ci; }
        Cb[idx] = f2bf(val);
      }
    }
  }
}

// ---------------------------------------------------------------------------
// softmax over the 8 slots for each (b, j) pixel column, + EPS
// ---------------------------------------------------------------------------
__global__ void softmax_slots_kernel(float* __restrict__ attn) {
  long id = (long)blockIdx.x * 256 + threadIdx.x;   // B*N = 131072
  int b = (int)(id >> 12), j = (int)(id & 4095);
  float* p = attn + ((long)b * 8) * 4096 + j;
  float v[8], mx = -1e30f;
#pragma unroll
  for (int i = 0; i < 8; i++) { v[i] = p[(long)i * 4096]; mx = fmaxf(mx, v[i]); }
  float s = 0.f;
#pragma unroll
  for (int i = 0; i < 8; i++) { v[i] = __expf(v[i] - mx); s += v[i]; }
  float inv = 1.f / s;
#pragma unroll
  for (int i = 0; i < 8; i++) p[(long)i * 4096] = v[i] * inv + 1e-8f;
}

// per (b,i) row sum over 4096 pixels
__global__ void rowsum_kernel(const float* __restrict__ attn, float* __restrict__ sums) {
  __shared__ float red[256];
  int row = blockIdx.x;                              // B*8 = 256
  const float* p = attn + (long)row * 4096;
  float s = 0.f;
  for (int j = threadIdx.x; j < 4096; j += 256) s += p[j];
  red[threadIdx.x] = s; __syncthreads();
  for (int off = 128; off > 0; off >>= 1) {
    if ((int)threadIdx.x < off) red[threadIdx.x] += red[threadIdx.x + off];
    __syncthreads();
  }
  if (threadIdx.x == 0) sums[row] = red[0];
}

// attn /= rowsum; also emit bf16 copy for the WMMA weighted-sum GEMMs
__global__ void attn_norm_kernel(float* __restrict__ attn, const float* __restrict__ sums,
                                 bf16_t* __restrict__ attn_bf) {
  long id = (long)blockIdx.x * 256 + threadIdx.x;   // B*8*4096
  int row = (int)(id >> 12);
  float v = attn[id] / sums[row];
  attn[id] = v;
  attn_bf[id] = f2bf(v);
}

// GRU combine: slots <- (1-z)*n + z*slots_prev   (256 rows x 256 cols)
__global__ void gru_kernel(const float* __restrict__ gx, const float* __restrict__ gh,
                           float* __restrict__ slots) {
  long id = (long)blockIdx.x * 256 + threadIdx.x;   // 65536
  int row = (int)(id >> 8), c = (int)(id & 255);
  const float* gxr = gx + (long)row * 768;
  const float* ghr = gh + (long)row * 768;
  float r = sigmoidf_(gxr[c] + ghr[c]);
  float z = sigmoidf_(gxr[256 + c] + ghr[256 + c]);
  float n = tanhf(gxr[512 + c] + r * ghr[512 + c]);
  slots[id] = (1.f - z) * n + z * slots[id];
}

// pos_up[b,i,p] = sum_j attn[b,i,j] * pos[j,p]   (P=4, tiny)
__global__ void pos_up_kernel(const float* __restrict__ attn, const float* __restrict__ pos,
                              float* __restrict__ out) {
  int b = blockIdx.x;                 // 32 blocks x 32 threads
  int i = threadIdx.x >> 2, p = threadIdx.x & 3;
  const float* ar = attn + ((long)b * 8 + i) * 4096;
  float s = 0.f;
  for (int j = 0; j < 4096; j++) s += ar[j] * pos[j * 4 + p];
  out[((long)b * 8 + i) * 4 + p] = s;
}

// ---------------------------------------------------------------------------
// Host orchestration
// ---------------------------------------------------------------------------
extern "C" void kernel_launch(void* const* d_in, const int* in_sizes, int n_in,
                              void* d_out, int out_size, void* d_ws, size_t ws_size,
                              hipStream_t stream) {
  (void)in_sizes; (void)n_in; (void)out_size; (void)ws_size;
  const int B = 32, N = 4096, D = 256, NS = 8, ITERS = 3;
  const long BN = (long)B * N;                 // 131072
  const float scale = 0.0625f;                 // D^-0.5

  const float* inputs = (const float*)d_in[0];
  const float* feats  = (const float*)d_in[1];
  const float* pos    = (const float*)d_in[2];
  const float* mu     = (const float*)d_in[4];
  const float* sigma  = (const float*)d_in[5];
  const float* noise  = (const float*)d_in[6];
  const float* Wq = (const float*)d_in[7];  const float* bq = (const float*)d_in[8];
  const float* Wk = (const float*)d_in[9];  const float* bk = (const float*)d_in[10];
  const float* Wv = (const float*)d_in[11]; const float* bv = (const float*)d_in[12];
  const float* wih = (const float*)d_in[13]; const float* whh = (const float*)d_in[14];
  const float* bih = (const float*)d_in[15]; const float* bhh = (const float*)d_in[16];
  const float* W1 = (const float*)d_in[17]; const float* b1 = (const float*)d_in[18];
  const float* W2 = (const float*)d_in[19]; const float* b2 = (const float*)d_in[20];
  const float* ln_in_w = (const float*)d_in[21]; const float* ln_in_b = (const float*)d_in[22];
  const float* ln_s_w  = (const float*)d_in[23]; const float* ln_s_b  = (const float*)d_in[24];
  const float* ln_f_w  = (const float*)d_in[25]; const float* ln_f_b  = (const float*)d_in[26];

  // --- workspace carve-up (256B aligned) ---
  char* wsp = (char*)d_ws; size_t off = 0;
  auto carve = [&](size_t bytes) -> void* {
    void* p = wsp + off; off = (off + bytes + 255) & ~(size_t)255; return p;
  };
  bf16_t* Wq_bf  = (bf16_t*)carve(256 * 256 * 2);
  bf16_t* Wk_bf  = (bf16_t*)carve(256 * 256 * 2);
  bf16_t* Wv_bf  = (bf16_t*)carve(256 * 256 * 2);
  bf16_t* W1_bf  = (bf16_t*)carve(256 * 256 * 2);
  bf16_t* W2_bf  = (bf16_t*)carve(256 * 256 * 2);
  bf16_t* wih_bf = (bf16_t*)carve(768 * 256 * 2);
  bf16_t* whh_bf = (bf16_t*)carve(768 * 256 * 2);
  float*  slots  = (float*) carve((size_t)B * NS * D * 4);
  bf16_t* sn_bf  = (bf16_t*)carve(256 * 256 * 2);
  bf16_t* q_bf   = (bf16_t*)carve(256 * 256 * 2);
  float*  attn   = (float*) carve((size_t)B * NS * N * 4);
  bf16_t* attn_bf= (bf16_t*)carve((size_t)B * NS * N * 2);
  float*  sums   = (float*) carve(256 * 4);
  float*  upd    = (float*) carve((size_t)B * NS * D * 4);
  bf16_t* upd_bf = (bf16_t*)carve(256 * 256 * 2);
  bf16_t* slots_bf=(bf16_t*)carve(256 * 256 * 2);
  float*  gx     = (float*) carve(256 * 768 * 4);
  float*  gh     = (float*) carve(256 * 768 * 4);
  bf16_t* ff_bf  = (bf16_t*)carve(256 * 256 * 2);
  bf16_t* h1_bf  = (bf16_t*)carve(256 * 256 * 2);
  bf16_t* x_bf   = (bf16_t*)carve((size_t)BN * D * 2);
  bf16_t* k_bf   = (bf16_t*)carve((size_t)BN * D * 2);   // (B, N, D)
  bf16_t* vT_bf  = (bf16_t*)carve((size_t)BN * D * 2);   // (B, D, N)  transposed
  bf16_t* fT_bf  = (bf16_t*)carve((size_t)BN * D * 2);   // (B, D, N)  transposed

  auto cvt = [&](const float* s, bf16_t* d, long n) {
    int blocks = (int)((n + 255) / 256); if (blocks > 4096) blocks = 4096;
    f2bf_kernel<<<blocks, 256, 0, stream>>>(s, d, n);
  };
  auto gemm = [&](const bf16_t* A, long sA, const bf16_t* Bm, long sB,
                  const float* bias, float alpha, float* Cf, long sCf, int accum,
                  bf16_t* Cb, long sCb, int cbt, int M, int Mvalid, int Nn, int Kk,
                  int relu, int batch) {
    dim3 g((unsigned)(M / 16), (unsigned)(Nn / 256), (unsigned)batch);
    gemm_wmma_kernel<<<g, 256, 0, stream>>>(A, sA, Bm, sB, bias, alpha,
                                            Cf, sCf, accum, Cb, sCb, cbt,
                                            Mvalid, Nn, Kk, relu);
  };

  // ---- prep: bf16 weights, transposed feats, slot init ----
  cvt(Wq, Wq_bf, 256 * 256);   cvt(Wk, Wk_bf, 256 * 256);   cvt(Wv, Wv_bf, 256 * 256);
  cvt(W1, W1_bf, 256 * 256);   cvt(W2, W2_bf, 256 * 256);
  cvt(wih, wih_bf, 768 * 256); cvt(whh, whh_bf, 768 * 256);
  {
    dim3 tg(4096 / 32, 256 / 32, B);
    transpose_bf16_kernel<<<tg, 256, 0, stream>>>(feats, fT_bf);
  }
  slots_init_kernel<<<(B * NS * D) / 256, 256, 0, stream>>>(mu, sigma, noise, slots);

  // ---- x = LN(inputs) -> bf16 ; k = x@Wk^T + bk ; v = (x@Wv^T + bv)^T ----
  layernorm_rows_kernel<<<(unsigned)(BN / 8), 256, 0, stream>>>(inputs, ln_in_w, ln_in_b, x_bf);
  gemm(x_bf, 0, Wk_bf, 0, bk, 1.f, nullptr, 0, 0, k_bf, 0, 0, (int)BN, (int)BN, D, D, 0, 1);
  gemm(x_bf, 0, Wv_bf, 0, bv, 1.f, nullptr, 0, 0, vT_bf, 0, /*cbt=*/1, (int)BN, (int)BN, D, D, 0, 1);

  // ---- iterations ----
  for (int it = 0; it < ITERS; ++it) {
    layernorm_rows_kernel<<<32, 256, 0, stream>>>(slots, ln_s_w, ln_s_b, sn_bf);
    gemm(sn_bf, 0, Wq_bf, 0, bq, 1.f, nullptr, 0, 0, q_bf, 0, 0, 256, 256, D, D, 0, 1);
    // dots = scale * q @ k^T   (batched, M padded 8->16)
    gemm(q_bf, (long)NS * D, k_bf, (long)N * D, nullptr, scale,
         attn, (long)NS * N, 0, nullptr, 0, 0, 16, NS, N, D, 0, B);
    softmax_slots_kernel<<<(unsigned)(BN / 256), 256, 0, stream>>>(attn);
    rowsum_kernel<<<B * NS, 256, 0, stream>>>(attn, sums);
    attn_norm_kernel<<<(unsigned)((long)B * NS * N / 256), 256, 0, stream>>>(attn, sums, attn_bf);
    // updates = attn @ vT^T   (batched, both operands contiguous in K=N)
    gemm(attn_bf, (long)NS * N, vT_bf, (long)D * N, nullptr, 1.f,
         upd, (long)NS * D, 0, nullptr, 0, 0, 16, NS, D, N, 0, B);
    // GRU
    cvt(upd, upd_bf, 256 * 256);
    cvt(slots, slots_bf, 256 * 256);
    gemm(upd_bf, 0, wih_bf, 0, bih, 1.f, gx, 0, 0, nullptr, 0, 0, 256, 256, 768, D, 0, 1);
    gemm(slots_bf, 0, whh_bf, 0, bhh, 1.f, gh, 0, 0, nullptr, 0, 0, 256, 256, 768, D, 0, 1);
    gru_kernel<<<256, 256, 0, stream>>>(gx, gh, slots);
    // FF: slots += relu(LN(slots)@W1^T + b1)@W2^T + b2
    layernorm_rows_kernel<<<32, 256, 0, stream>>>(slots, ln_f_w, ln_f_b, ff_bf);
    gemm(ff_bf, 0, W1_bf, 0, b1, 1.f, nullptr, 0, 0, h1_bf, 0, 0, 256, 256, D, D, 1, 1);
    gemm(h1_bf, 0, W2_bf, 0, b2, 1.f, slots, 0, 1, nullptr, 0, 0, 256, 256, D, D, 0, 1);
  }

  // ---- outputs: [slots | feat_up | pos_up | attn] ----
  float* out = (float*)d_out;
  gemm(attn_bf, (long)NS * N, fT_bf, (long)D * N, nullptr, 1.f,
       out + 65536, (long)NS * D, 0, nullptr, 0, 0, 16, NS, D, N, 0, B);
  pos_up_kernel<<<B, 32, 0, stream>>>(attn, pos, out + 131072);
  hipMemcpyAsync(out, slots, (size_t)B * NS * D * 4, hipMemcpyDeviceToDevice, stream);
  hipMemcpyAsync(out + 132096, attn, (size_t)B * NS * N * 4, hipMemcpyDeviceToDevice, stream);
}
